// unoNet_88270167867639
// MI455X (gfx1250) — compile-verified, matrix operand order checked by reading, not run on hardware
//
#include <hip/hip_runtime.h>
#include <hip/hip_bf16.h>
#include <math.h>

typedef __bf16 bf16_t;
typedef bf16_t v16bf __attribute__((ext_vector_type(16)));
typedef float  v8f   __attribute__((ext_vector_type(8)));

#define P_N   20000
#define L_N   16
#define NL_N  50000
#define NN_N  4096
#define DS_N  32      // DL == DN == DP == 32
#define H_N   128
#define RU_N  256

// ---------- WMMA helpers (gfx1250 layouts, wave32) ----------

// K-pair base for 16-bit A/B fragment VGPR v, lane-half h.
__device__ __forceinline__ int a_kbase(int v, int hf) {
  return ((v & 4) << 2) + (hf << 3) + ((v & 3) << 1);
}

__device__ __forceinline__ v8f v8zero() {
  v8f z = {0.f, 0.f, 0.f, 0.f, 0.f, 0.f, 0.f, 0.f};
  return z;
}

__device__ __forceinline__ v8f wmma_bf16(v16bf a, v16bf b, v8f c) {
  return __builtin_amdgcn_wmma_f32_16x16x32_bf16(false, a, false, b, (short)0, c,
                                                 false, false);
}

// A fragment (16x32) from 32 contiguous fp32 features of this lane's row.
__device__ __forceinline__ v16bf load_a_f32row(const float* __restrict__ row, int hf) {
  v16bf a;
#pragma unroll
  for (int v = 0; v < 8; ++v) {
    int kb = a_kbase(v, hf);
    a[2 * v]     = (bf16_t)row[kb];
    a[2 * v + 1] = (bf16_t)row[kb + 1];
  }
  return a;
}

// A fragment from bf16 LDS tile, row-major with leading dim ld, K offset k0.
__device__ __forceinline__ v16bf load_a_bf16(const bf16_t* base, int ld, int k0,
                                             int mrow, int hf) {
  v16bf a;
#pragma unroll
  for (int v = 0; v < 8; ++v) {
    int kb = k0 + a_kbase(v, hf);
    a[2 * v]     = base[mrow * ld + kb];
    a[2 * v + 1] = base[mrow * ld + kb + 1];
  }
  return a;
}

// B fragment (32x16) from bf16 KxN row-major weights; caller passes src
// already offset to the K-chunk start; col = tile_n*16 + (lane&15).
__device__ __forceinline__ v16bf load_b(const bf16_t* __restrict__ src, int ld,
                                        int col, int hf) {
  v16bf b;
#pragma unroll
  for (int v = 0; v < 8; ++v) {
    int kb = a_kbase(v, hf);
    b[2 * v]     = src[kb * ld + col];
    b[2 * v + 1] = src[(kb + 1) * ld + col];
  }
  return b;
}

__device__ __forceinline__ float sigmoidf_(float x) {
  return 1.0f / (1.0f + __expf(-x));
}

// ---------- utility kernels ----------

__global__ void k_cvt_bf16(const float* __restrict__ src, bf16_t* __restrict__ dst,
                           int n) {
  int i = blockIdx.x * blockDim.x + threadIdx.x;
  if (i < n) dst[i] = (bf16_t)src[i];
}

__global__ void k_zero(float* __restrict__ p, int n) {
  int i = blockIdx.x * blockDim.x + threadIdx.x;
  if (i < n) p[i] = 0.0f;
}

// state[r][c] = (c==0) ? vals[r] : 0
__global__ void k_init_col0(const float* __restrict__ vals, float* __restrict__ st,
                            int n) {
  int i = blockIdx.x * blockDim.x + threadIdx.x;
  if (i >= n) return;
  int r = i >> 5, c = i & 31;
  st[i] = (c == 0) ? vals[r] : 0.0f;
}

// path_state[p][c] = c==0 ? traffic[0][p] : c==1 ? traffic[1][p] : 0
__global__ void k_init_path(const float* __restrict__ traffic,
                            float* __restrict__ ps, int n) {
  int i = blockIdx.x * blockDim.x + threadIdx.x;
  if (i >= n) return;
  int p = i >> 5, c = i & 31;
  ps[i] = (c == 0) ? traffic[p] : (c == 1) ? traffic[P_N + p] : 0.0f;
}

// ---------- GRU over L=16 steps, fused incidence gather + message scatter ----------
#define GRU_NW 2
__global__ void __launch_bounds__(64) k_gru(
    const float* __restrict__ link_state, const float* __restrict__ node_state,
    float* __restrict__ path_state, float* __restrict__ m_acc,
    const int* __restrict__ links_to_paths, const int* __restrict__ nodes_to_paths,
    const bf16_t* __restrict__ Wk /*64x96*/, const bf16_t* __restrict__ Wrk /*32x96*/,
    const float* __restrict__ gbias /*2x96*/) {
  __shared__ bf16_t hstage[GRU_NW][16 * DS_N];
  const int wave = threadIdx.x >> 5, lane = threadIdx.x & 31;
  const int p0 = (blockIdx.x * GRU_NW + wave) * 16;
  const int mrow = lane & 15, hf = lane >> 4, n = lane & 15;
  bf16_t* hs = &hstage[wave][0];

  // h0 = path_state tile, in C layout; also stage as bf16 A tile.
  v8f hc[2];
#pragma unroll
  for (int ct = 0; ct < 2; ++ct)
#pragma unroll
    for (int i = 0; i < 8; ++i) {
      float h = path_state[(size_t)(p0 + i + hf * 8) * DS_N + ct * 16 + n];
      hc[ct][i] = h;
      hs[(i + hf * 8) * DS_N + ct * 16 + n] = (bf16_t)h;
    }
  __syncthreads();

  for (int t = 0; t < L_N; ++t) {
    const int e = (p0 + mrow) * L_N + t;
    const float* lsrc = link_state + (size_t)links_to_paths[e] * DS_N;
    const float* nsrc = node_state + (size_t)nodes_to_paths[e] * DS_N;
    v16bf ax0 = load_a_f32row(lsrc, hf);          // x features 0..31  (link)
    v16bf ax1 = load_a_f32row(nsrc, hf);          // x features 32..63 (node)
    v16bf ah  = load_a_bf16(hs, DS_N, 0, mrow, hf);

    v8f mx[6], mh[6];
#pragma unroll
    for (int j = 0; j < 6; ++j) {
      int col = j * 16 + n;
      v16bf b0 = load_b(Wk, 96, col, hf);
      v16bf b1 = load_b(Wk + 32 * 96, 96, col, hf);
      v16bf br = load_b(Wrk, 96, col, hf);
      v8f ax = v8zero();
      ax = wmma_bf16(ax0, b0, ax);
      ax = wmma_bf16(ax1, b1, ax);
      v8f ahh = v8zero();
      ahh = wmma_bf16(ah, br, ahh);
      float bx = gbias[col], bh = gbias[96 + col];
#pragma unroll
      for (int i = 0; i < 8; ++i) { ax[i] += bx; ahh[i] += bh; }
      mx[j] = ax;
      mh[j] = ahh;
    }

    __syncthreads();  // all reads of hs for this step done
#pragma unroll
    for (int ct = 0; ct < 2; ++ct) {
#pragma unroll
      for (int i = 0; i < 8; ++i) {
        float z  = sigmoidf_(mx[ct][i] + mh[ct][i]);
        float r  = sigmoidf_(mx[2 + ct][i] + mh[2 + ct][i]);
        float cc = tanhf(mx[4 + ct][i] + r * mh[4 + ct][i]);
        float hn = z * hc[ct][i] + (1.0f - z) * cc;
        hc[ct][i] = hn;
        int row = i + hf * 8;
        hs[row * DS_N + ct * 16 + n] = (bf16_t)hn;
        int lk = links_to_paths[(p0 + row) * L_N + t];
        atomicAdd(&m_acc[(size_t)lk * DS_N + ct * 16 + n], hn);
      }
    }
    __syncthreads();
  }

#pragma unroll
  for (int ct = 0; ct < 2; ++ct)
#pragma unroll
    for (int i = 0; i < 8; ++i)
      path_state[(size_t)(p0 + i + hf * 8) * DS_N + ct * 16 + n] = hc[ct][i];
}

// ---------- edge MLP: relu(relu(con@We1)@We2)@We3 -> link_state ----------
#define EDGE_NW 5
__global__ void __launch_bounds__(160) k_edge(
    const float* __restrict__ node_state, float* __restrict__ link_state,
    const float* __restrict__ m_acc, const int* __restrict__ links_to_nodes,
    const bf16_t* __restrict__ We1, const float* __restrict__ be1,
    const bf16_t* __restrict__ We2, const float* __restrict__ be2,
    const bf16_t* __restrict__ We3, const float* __restrict__ be3) {
  __shared__ bf16_t st1[EDGE_NW][16 * H_N];
  __shared__ bf16_t st2[EDGE_NW][16 * H_N];
  const int wave = threadIdx.x >> 5, lane = threadIdx.x & 31;
  const int l0 = (blockIdx.x * EDGE_NW + wave) * 16;
  const int mrow = lane & 15, hf = lane >> 4, n = lane & 15;
  bf16_t* s1 = &st1[wave][0];
  bf16_t* s2 = &st2[wave][0];

  // con = [node_state[links_to_nodes[l]] | link_state[l] | m[l]]  (16x96)
  const float* sn = node_state + (size_t)links_to_nodes[l0 + mrow] * DS_N;
  v16bf aN = load_a_f32row(sn, hf);
  v16bf aL = load_a_f32row(link_state + (size_t)(l0 + mrow) * DS_N, hf);
  v16bf aM = load_a_f32row(m_acc + (size_t)(l0 + mrow) * DS_N, hf);

  // layer 1: 16x96 @ 96x128, relu
#pragma unroll
  for (int j = 0; j < 8; ++j) {
    int col = j * 16 + n;
    v8f acc = v8zero();
    acc = wmma_bf16(aN, load_b(We1, H_N, col, hf), acc);
    acc = wmma_bf16(aL, load_b(We1 + 32 * H_N, H_N, col, hf), acc);
    acc = wmma_bf16(aM, load_b(We1 + 64 * H_N, H_N, col, hf), acc);
    float b = be1[col];
#pragma unroll
    for (int i = 0; i < 8; ++i)
      s1[(i + hf * 8) * H_N + col] = (bf16_t)fmaxf(acc[i] + b, 0.0f);
  }
  __syncthreads();

  v16bf x1[4];
#pragma unroll
  for (int c = 0; c < 4; ++c) x1[c] = load_a_bf16(s1, H_N, c * 32, mrow, hf);

  // layer 2: 16x128 @ 128x128, relu
#pragma unroll
  for (int j = 0; j < 8; ++j) {
    int col = j * 16 + n;
    v8f acc = v8zero();
#pragma unroll
    for (int c = 0; c < 4; ++c)
      acc = wmma_bf16(x1[c], load_b(We2 + c * 32 * H_N, H_N, col, hf), acc);
    float b = be2[col];
#pragma unroll
    for (int i = 0; i < 8; ++i)
      s2[(i + hf * 8) * H_N + col] = (bf16_t)fmaxf(acc[i] + b, 0.0f);
  }
  __syncthreads();

  v16bf x2[4];
#pragma unroll
  for (int c = 0; c < 4; ++c) x2[c] = load_a_bf16(s2, H_N, c * 32, mrow, hf);

  // layer 3: 16x128 @ 128x32 -> new link_state
#pragma unroll
  for (int j = 0; j < 2; ++j) {
    int col = j * 16 + n;
    v8f acc = v8zero();
#pragma unroll
    for (int c = 0; c < 4; ++c)
      acc = wmma_bf16(x2[c], load_b(We3 + c * 32 * DS_N, DS_N, col, hf), acc);
    float b = be3[col];
#pragma unroll
    for (int i = 0; i < 8; ++i)
      link_state[(size_t)(l0 + i + hf * 8) * DS_N + col] = acc[i] + b;
  }
}

// ---------- node aggregation: agg[links_to_nodes[l]] += link_state[nodes_to_links[l]]
__global__ void k_node_agg(const float* __restrict__ link_state,
                           float* __restrict__ agg,
                           const int* __restrict__ links_to_nodes,
                           const int* __restrict__ nodes_to_links) {
  int i = blockIdx.x * blockDim.x + threadIdx.x;  // NL*32 threads
  int l = i >> 5, c = i & 31;
  atomicAdd(&agg[(size_t)links_to_nodes[l] * DS_N + c],
            link_state[(size_t)nodes_to_links[l] * DS_N + c]);
}

// ---------- GCN projection: tmpb = (bf16)([node_state|agg] @ Wg)  (NN x 32) ----------
#define GP_NW 4
__global__ void __launch_bounds__(128) k_gcn_pre(
    const float* __restrict__ node_state, const float* __restrict__ agg,
    const bf16_t* __restrict__ Wg /*64x32*/, bf16_t* __restrict__ tmpb) {
  const int wave = threadIdx.x >> 5, lane = threadIdx.x & 31;
  const int r0 = (blockIdx.x * GP_NW + wave) * 16;
  const int mrow = lane & 15, hf = lane >> 4, n = lane & 15;
  v16bf aN = load_a_f32row(node_state + (size_t)(r0 + mrow) * DS_N, hf);
  v16bf aA = load_a_f32row(agg + (size_t)(r0 + mrow) * DS_N, hf);
#pragma unroll
  for (int j = 0; j < 2; ++j) {
    int col = j * 16 + n;
    v8f acc = v8zero();
    acc = wmma_bf16(aN, load_b(Wg, DS_N, col, hf), acc);
    acc = wmma_bf16(aA, load_b(Wg + 32 * DS_N, DS_N, col, hf), acc);
#pragma unroll
    for (int i = 0; i < 8; ++i)
      tmpb[(size_t)(r0 + i + hf * 8) * DS_N + col] = (bf16_t)acc[i];
  }
}

// ---------- GCN main: node_state = laplacian(NNxNN) @ tmp(NNx32) + bg ----------
#define GM_NW 4
__global__ void __launch_bounds__(128) k_gcn_main(
    const float* __restrict__ lap, const bf16_t* __restrict__ tmpb,
    float* __restrict__ node_state, const float* __restrict__ bg) {
  const int wave = threadIdx.x >> 5, lane = threadIdx.x & 31;
  const int r0 = (blockIdx.x * GM_NW + wave) * 16;
  const int mrow = lane & 15, hf = lane >> 4, n = lane & 15;
  v8f acc0 = v8zero(), acc1 = v8zero();
  const float* arow = lap + (size_t)(r0 + mrow) * NN_N;
  for (int kk = 0; kk < NN_N / 32; ++kk) {
    __builtin_prefetch(arow + kk * 32 + 64, 0, 1);   // global_prefetch_b8
    v16bf a = load_a_f32row(arow + kk * 32, hf);
    const bf16_t* bsrc = tmpb + (size_t)kk * 32 * DS_N;
    acc0 = wmma_bf16(a, load_b(bsrc, DS_N, n, hf), acc0);
    acc1 = wmma_bf16(a, load_b(bsrc, DS_N, 16 + n, hf), acc1);
  }
  float b0 = bg[n], b1 = bg[16 + n];
#pragma unroll
  for (int i = 0; i < 8; ++i) {
    size_t row = (size_t)(r0 + i + hf * 8) * DS_N;
    node_state[row + n]      = acc0[i] + b0;
    node_state[row + 16 + n] = acc1[i] + b1;
  }
}

// ---------- readout: relu(relu(ps@Wr1)@Wr2); out = [r|ps]@Wf + bf ----------
#define RO_NW 2
__global__ void __launch_bounds__(64) k_readout(
    const float* __restrict__ path_state, const bf16_t* __restrict__ Wr1,
    const float* __restrict__ br1, const bf16_t* __restrict__ Wr2,
    const float* __restrict__ br2, const float* __restrict__ Wf,
    const float* __restrict__ bfv, float* __restrict__ out) {
  __shared__ bf16_t r1s[RO_NW][16 * RU_N];
  __shared__ bf16_t r2s[RO_NW][16 * RU_N];
  const int wave = threadIdx.x >> 5, lane = threadIdx.x & 31;
  const int p0 = (blockIdx.x * RO_NW + wave) * 16;
  const int mrow = lane & 15, hf = lane >> 4, n = lane & 15;
  bf16_t* s1 = &r1s[wave][0];
  bf16_t* s2 = &r2s[wave][0];

  v16bf aP = load_a_f32row(path_state + (size_t)(p0 + mrow) * DS_N, hf);
  for (int j = 0; j < 16; ++j) {
    int col = j * 16 + n;
    v8f acc = v8zero();
    acc = wmma_bf16(aP, load_b(Wr1, RU_N, col, hf), acc);
    float b = br1[col];
#pragma unroll
    for (int i = 0; i < 8; ++i)
      s1[(i + hf * 8) * RU_N + col] = (bf16_t)fmaxf(acc[i] + b, 0.0f);
  }
  __syncthreads();

  v16bf r1a[8];
#pragma unroll
  for (int c = 0; c < 8; ++c) r1a[c] = load_a_bf16(s1, RU_N, c * 32, mrow, hf);

  for (int j = 0; j < 16; ++j) {
    int col = j * 16 + n;
    v8f acc = v8zero();
#pragma unroll
    for (int c = 0; c < 8; ++c)
      acc = wmma_bf16(r1a[c], load_b(Wr2 + c * 32 * RU_N, RU_N, col, hf), acc);
    float b = br2[col];
#pragma unroll
    for (int i = 0; i < 8; ++i)
      s2[(i + hf * 8) * RU_N + col] = (bf16_t)fmaxf(acc[i] + b, 0.0f);
  }
  __syncthreads();

  if (lane < 16) {
    int p = p0 + lane;
    float acc = bfv[0];
    for (int k = 0; k < RU_N; ++k) acc += (float)s2[lane * RU_N + k] * Wf[k];
    for (int k = 0; k < DS_N; ++k)
      acc += path_state[(size_t)p * DS_N + k] * Wf[RU_N + k];
    out[p] = acc;
  }
}

// ---------- host orchestration ----------

extern "C" void kernel_launch(void* const* d_in, const int* in_sizes, int n_in,
                              void* d_out, int out_size, void* d_ws, size_t ws_size,
                              hipStream_t stream) {
  const float* capacities = (const float*)d_in[0];
  const float* degrees    = (const float*)d_in[1];
  const float* traffic    = (const float*)d_in[2];
  const float* lap        = (const float*)d_in[3];
  const int* links_to_paths = (const int*)d_in[6];
  const int* nodes_to_paths = (const int*)d_in[7];
  const int* links_to_nodes = (const int*)d_in[8];
  const int* nodes_to_links = (const int*)d_in[9];
  const float* gru_k  = (const float*)d_in[13];
  const float* gru_rk = (const float*)d_in[14];
  const float* gru_b  = (const float*)d_in[15];
  const float* We1 = (const float*)d_in[16]; const float* be1 = (const float*)d_in[17];
  const float* We2 = (const float*)d_in[18]; const float* be2 = (const float*)d_in[19];
  const float* We3 = (const float*)d_in[20]; const float* be3 = (const float*)d_in[21];
  const float* Wg  = (const float*)d_in[22]; const float* bg  = (const float*)d_in[23];
  const float* Wr1 = (const float*)d_in[24]; const float* br1 = (const float*)d_in[25];
  const float* Wr2 = (const float*)d_in[26]; const float* br2 = (const float*)d_in[27];
  const float* Wf  = (const float*)d_in[28]; const float* bfv = (const float*)d_in[29];

  char* ws = (char*)d_ws;
  size_t off = 0;
  auto alloc = [&](size_t bytes) -> void* {
    void* p = ws + off;
    off = (off + bytes + 255) & ~(size_t)255;
    return p;
  };

  float* link_state = (float*)alloc((size_t)NL_N * DS_N * 4);
  float* node_state = (float*)alloc((size_t)NN_N * DS_N * 4);
  float* path_state = (float*)alloc((size_t)P_N * DS_N * 4);
  float* m_acc      = (float*)alloc((size_t)NL_N * DS_N * 4);
  float* agg        = (float*)alloc((size_t)NN_N * DS_N * 4);
  bf16_t* tmpb      = (bf16_t*)alloc((size_t)NN_N * DS_N * 2);
  bf16_t* wkb  = (bf16_t*)alloc(64 * 96 * 2);
  bf16_t* wrkb = (bf16_t*)alloc(32 * 96 * 2);
  bf16_t* we1b = (bf16_t*)alloc(96 * 128 * 2);
  bf16_t* we2b = (bf16_t*)alloc(128 * 128 * 2);
  bf16_t* we3b = (bf16_t*)alloc(128 * 32 * 2);
  bf16_t* wgb  = (bf16_t*)alloc(64 * 32 * 2);
  bf16_t* wr1b = (bf16_t*)alloc(32 * 256 * 2);
  bf16_t* wr2b = (bf16_t*)alloc(256 * 256 * 2);

  // weight conversion to bf16
  auto cvt = [&](const float* s, bf16_t* d, int n) {
    k_cvt_bf16<<<(n + 255) / 256, 256, 0, stream>>>(s, d, n);
  };
  cvt(gru_k, wkb, 64 * 96);   cvt(gru_rk, wrkb, 32 * 96);
  cvt(We1, we1b, 96 * 128);   cvt(We2, we2b, 128 * 128);
  cvt(We3, we3b, 128 * 32);   cvt(Wg, wgb, 64 * 32);
  cvt(Wr1, wr1b, 32 * 256);   cvt(Wr2, wr2b, 256 * 256);

  // initial states
  k_init_col0<<<(NL_N * DS_N) / 256, 256, 0, stream>>>(capacities, link_state,
                                                       NL_N * DS_N);
  k_init_col0<<<(NN_N * DS_N) / 256, 256, 0, stream>>>(degrees, node_state,
                                                       NN_N * DS_N);
  k_init_path<<<(P_N * DS_N) / 256, 256, 0, stream>>>(traffic, path_state,
                                                      P_N * DS_N);

  for (int it = 0; it < 4; ++it) {
    k_zero<<<(NL_N * DS_N) / 256, 256, 0, stream>>>(m_acc, NL_N * DS_N);
    k_gru<<<(P_N / 16) / GRU_NW, GRU_NW * 32, 0, stream>>>(
        link_state, node_state, path_state, m_acc, links_to_paths, nodes_to_paths,
        wkb, wrkb, gru_b);
    k_edge<<<(NL_N / 16) / EDGE_NW, EDGE_NW * 32, 0, stream>>>(
        node_state, link_state, m_acc, links_to_nodes, we1b, be1, we2b, be2, we3b,
        be3);
    k_zero<<<(NN_N * DS_N) / 256, 256, 0, stream>>>(agg, NN_N * DS_N);
    k_node_agg<<<(NL_N * DS_N) / 256, 256, 0, stream>>>(link_state, agg,
                                                        links_to_nodes,
                                                        nodes_to_links);
    k_gcn_pre<<<(NN_N / 16) / GP_NW, GP_NW * 32, 0, stream>>>(node_state, agg, wgb,
                                                              tmpb);
    k_gcn_main<<<(NN_N / 16) / GM_NW, GM_NW * 32, 0, stream>>>(lap, tmpb,
                                                               node_state, bg);
  }

  k_readout<<<(P_N / 16) / RO_NW, RO_NW * 32, 0, stream>>>(
      path_state, wr1b, br1, wr2b, br2, Wf, bfv, (float*)d_out);
}
